// SelfAttention_3032246911366
// MI455X (gfx1250) — matrix operand order repaired and checked by reading, not compiled, not run on hardware
//
#include <hip/hip_runtime.h>

// ---------------------------------------------------------------------------
// Types / helpers for CDNA5 WMMA (gfx1250, wave32)
// ---------------------------------------------------------------------------
typedef __attribute__((ext_vector_type(16))) __bf16 v16bf;
typedef __attribute__((ext_vector_type(8)))  float  v8f;

struct __align__(16) U128 { unsigned x, y, z, w; };
union FragU { U128 q[2]; v16bf v; };

__device__ __forceinline__ v16bf make_frag(U128 a, U128 b) {
  FragU u; u.q[0] = a; u.q[1] = b; return u.v;
}
// 16 contiguous bf16 (32 bytes, 16B aligned)
__device__ __forceinline__ v16bf frag16(const unsigned short* p) {
  const U128* q = (const U128*)p;
  return make_frag(q[0], q[1]);
}
// A-fragment: 16 contiguous elements at +hi*16 (same k order as B frag16)
__device__ __forceinline__ v16bf fragA(const unsigned short* row, unsigned hi) {
  return make_frag(*(const U128*)(row + hi * 16), *(const U128*)(row + hi * 16 + 8));
}
__device__ __forceinline__ unsigned short f2bf(float f) {
  unsigned u = __builtin_bit_cast(unsigned, f);
  u += 0x7FFFu + ((u >> 16) & 1u);   // round-to-nearest-even
  return (unsigned short)(u >> 16);
}

#define WMMA_BF16(A, B, C) \
  __builtin_amdgcn_wmma_f32_16x16x32_bf16(false, (A), false, (B), (short)0, (C), false, false)

// Problem constants
#define EMBED 1024
#define SEQ   2048
#define NB    4
#define NH    16
#define HD    64
#define MROWS (NB * SEQ)   // 8192

// ---------------------------------------------------------------------------
// Conversion kernels
// ---------------------------------------------------------------------------
__global__ void cvt_f32_to_bf16(const float* __restrict__ in,
                                unsigned short* __restrict__ out, int n) {
  int i = blockIdx.x * blockDim.x + threadIdx.x;
  if (i < n) out[i] = f2bf(in[i]);
}

// WT[n][k] = W[k][n]  (bf16, so B-fragment columns are contiguous)
__global__ void cvt_transpose_bf16(const float* __restrict__ W,
                                   unsigned short* __restrict__ WT) {
  int i = blockIdx.x * blockDim.x + threadIdx.x;   // i = n*1024 + k
  int n = i >> 10, k = i & 1023;
  WT[i] = f2bf(W[k * EMBED + n]);
}

// ---------------------------------------------------------------------------
// 64x64 register-blocked bf16 GEMM body: 4 A-frags x 4 B-frags ->
// 16 independent WMMAs per 32-wide K step (1 b128 load per WMMA).
// ---------------------------------------------------------------------------
__device__ __forceinline__ void gemm_64x64(const unsigned short* __restrict__ A,
                                           const unsigned short* __restrict__ BT,
                                           int m0, int n0, unsigned r, unsigned hi,
                                           v8f acc[4][4]) {
  const unsigned short* aRow[4];
  const unsigned short* bCol[4];
#pragma unroll
  for (int i = 0; i < 4; ++i) {
    aRow[i] = A  + (size_t)(m0 + i * 16 + r) * EMBED;
    bCol[i] = BT + (size_t)(n0 + i * 16 + r) * EMBED;
  }
  for (int k0 = 0; k0 < EMBED; k0 += 32) {
    v16bf af[4], bf[4];
#pragma unroll
    for (int i = 0; i < 4; ++i) af[i] = fragA(aRow[i] + k0, hi);
#pragma unroll
    for (int j = 0; j < 4; ++j) bf[j] = frag16(bCol[j] + k0 + hi * 16);
#pragma unroll
    for (int i = 0; i < 4; ++i)
#pragma unroll
      for (int j = 0; j < 4; ++j)
        acc[i][j] = WMMA_BF16(af[i], bf[j], acc[i][j]);
  }
}

// ---------------------------------------------------------------------------
// Fused Q/K/V projection: Y = x @ W + b. One wave = one 64x64 output tile.
// grid.z selects projection. V stored transposed per (b,h): Vt[b][h][d][s].
// ---------------------------------------------------------------------------
__global__ void __launch_bounds__(256, 1) qkv_gemm(
    const unsigned short* __restrict__ xb,
    const unsigned short* __restrict__ wqT, const unsigned short* __restrict__ wkT,
    const unsigned short* __restrict__ wvT,
    const float* __restrict__ bq, const float* __restrict__ bk, const float* __restrict__ bv,
    unsigned short* __restrict__ Qb, unsigned short* __restrict__ Kb,
    unsigned short* __restrict__ Vt)
{
  const unsigned lane = threadIdx.x & 31u, wid = threadIdx.x >> 5;
  const unsigned r = lane & 15u, hi = lane >> 4;
  const int proj = blockIdx.z;
  const int m0 = (blockIdx.y * 8 + wid) * 64;   // 128 M tiles of 64
  const int n0 = blockIdx.x * 64;               // 16  N tiles of 64

  const unsigned short* wT   = (proj == 0) ? wqT : ((proj == 1) ? wkT : wvT);
  const float*          bias = (proj == 0) ? bq  : ((proj == 1) ? bk  : bv);

  v8f acc[4][4] = {};
  gemm_64x64(xb, wT, m0, n0, r, hi, acc);

  float bs[4];
#pragma unroll
  for (int j = 0; j < 4; ++j) bs[j] = bias[n0 + j * 16 + r];

  if (proj == 2) {
#pragma unroll
    for (int i = 0; i < 4; ++i)
      for (int j = 0; j < 4; ++j)
        for (int v = 0; v < 8; ++v) {
          const int mrow = m0 + i * 16 + v + 8 * hi;
          const int e = n0 + j * 16 + r, hh = e >> 6, dd = e & 63;
          const int bb = mrow >> 11, sl = mrow & 2047;
          Vt[(((size_t)bb * NH + hh) * HD + dd) * SEQ + sl] = f2bf(acc[i][j][v] + bs[j]);
        }
  } else {
    unsigned short* out = (proj == 0) ? Qb : Kb;
#pragma unroll
    for (int i = 0; i < 4; ++i)
      for (int j = 0; j < 4; ++j)
        for (int v = 0; v < 8; ++v)
          out[(size_t)(m0 + i * 16 + v + 8 * hi) * EMBED + n0 + j * 16 + r] =
              f2bf(acc[i][j][v] + bs[j]);
  }
}

// ---------------------------------------------------------------------------
// Flash attention: one wave per (b, h, 32-row query tile). Two 16-row query
// subtiles share every K and V fragment (halves K/V L2 traffic). Keys in
// blocks of 32 with online softmax; P converts C->A layout through LDS.
// ---------------------------------------------------------------------------
__global__ void __launch_bounds__(256, 1) flash_attn(
    const unsigned short* __restrict__ Qb, const unsigned short* __restrict__ Kb,
    const unsigned short* __restrict__ Vt, unsigned short* __restrict__ Ob)
{
  __shared__ __align__(16) unsigned short Pt[8][2][16 * 32];  // per wave, per q-subtile
  const unsigned lane = threadIdx.x & 31u, wid = threadIdx.x >> 5;
  const unsigned r = lane & 15u, hi = lane >> 4;

  const int id = blockIdx.x * 8 + wid;          // 0..4095
  const int qt = id & 63;                       // 32-row query tile
  const int h  = (id >> 6) & 15;
  const int b  = id >> 10;

  // Q A-fragments: [q-subtile][k-chunk of 32]
  v16bf qf[2][2];
#pragma unroll
  for (int qs = 0; qs < 2; ++qs) {
    const unsigned short* qRow =
        Qb + (size_t)(b * SEQ + qt * 32 + qs * 16 + r) * EMBED + h * HD;
    qf[qs][0] = fragA(qRow, hi);
    qf[qs][1] = fragA(qRow + 32, hi);
  }

  const unsigned short* kBase = Kb + (size_t)(b * SEQ) * EMBED + h * HD;
  const unsigned short* vBase = Vt + (size_t)(b * NH + h) * HD * SEQ;

  float m_i[2][8], l_i[2][8];
  v8f o[2][4] = {};
#pragma unroll
  for (int qs = 0; qs < 2; ++qs)
    for (int v = 0; v < 8; ++v) { m_i[qs][v] = -3.0e38f; l_i[qs][v] = 0.0f; }

  for (int kb = 0; kb < SEQ; kb += 32) {
    const unsigned short* kRow0 = kBase + (size_t)(kb + r) * EMBED;
    const unsigned short* kRow1 = kRow0 + (size_t)16 * EMBED;
    // Prefetch next key block's rows (global_prefetch_b8)
    if (kb + 32 < SEQ) {
      __builtin_prefetch(kRow0 + (size_t)32 * EMBED, 0, 1);
      __builtin_prefetch(kRow1 + (size_t)32 * EMBED, 0, 1);
    }
    // K B-fragments shared by both query subtiles
    const v16bf kf0a = frag16(kRow0 + hi * 16);
    const v16bf kf0b = frag16(kRow0 + 32 + hi * 16);
    const v16bf kf1a = frag16(kRow1 + hi * 16);
    const v16bf kf1b = frag16(kRow1 + 32 + hi * 16);

#pragma unroll
    for (int qs = 0; qs < 2; ++qs) {
      v8f s0 = {}, s1 = {};
      s0 = WMMA_BF16(qf[qs][0], kf0a, s0);
      s0 = WMMA_BF16(qf[qs][1], kf0b, s0);
      s1 = WMMA_BF16(qf[qs][0], kf1a, s1);
      s1 = WMMA_BF16(qf[qs][1], kf1b, s1);

      float alpha[8];
#pragma unroll
      for (int v = 0; v < 8; ++v) {
        const float a = s0[v] * 0.125f;
        const float c = s1[v] * 0.125f;
        float t = fmaxf(a, c);
        t = fmaxf(t, __shfl_xor(t, 1, 32));
        t = fmaxf(t, __shfl_xor(t, 2, 32));
        t = fmaxf(t, __shfl_xor(t, 4, 32));
        t = fmaxf(t, __shfl_xor(t, 8, 32));
        const float mn = fmaxf(m_i[qs][v], t);
        alpha[v] = __expf(m_i[qs][v] - mn);
        m_i[qs][v] = mn;
        const float p0 = __expf(a - mn);
        const float p1 = __expf(c - mn);
        float rs = p0 + p1;
        rs += __shfl_xor(rs, 1, 32);
        rs += __shfl_xor(rs, 2, 32);
        rs += __shfl_xor(rs, 4, 32);
        rs += __shfl_xor(rs, 8, 32);
        l_i[qs][v] = l_i[qs][v] * alpha[v] + rs;
        const int row = v + 8 * hi;           // C-layout row
        Pt[wid][qs][row * 32 + r]      = f2bf(p0);
        Pt[wid][qs][row * 32 + 16 + r] = f2bf(p1);
        o[qs][0][v] *= alpha[v]; o[qs][1][v] *= alpha[v];
        o[qs][2][v] *= alpha[v]; o[qs][3][v] *= alpha[v];
      }
    }
    __syncthreads();
    v16bf pf[2];
#pragma unroll
    for (int qs = 0; qs < 2; ++qs)
      pf[qs] = fragA(&Pt[wid][qs][r * 32], hi);
    __syncthreads();

    // V B-fragments shared by both query subtiles; 8 PV WMMAs
#pragma unroll
    for (int c = 0; c < 4; ++c) {
      const v16bf vf = frag16(vBase + (size_t)(c * 16 + r) * SEQ + kb + hi * 16);
      o[0][c] = WMMA_BF16(pf[0], vf, o[0][c]);
      o[1][c] = WMMA_BF16(pf[1], vf, o[1][c]);
    }
  }

#pragma unroll
  for (int qs = 0; qs < 2; ++qs)
    for (int v = 0; v < 8; ++v) {
      const float inv = 1.0f / l_i[qs][v];
      const size_t base =
          (size_t)(b * SEQ + qt * 32 + qs * 16 + v + 8 * hi) * EMBED + h * HD + r;
      Ob[base + 0]  = f2bf(o[qs][0][v] * inv);
      Ob[base + 16] = f2bf(o[qs][1][v] * inv);
      Ob[base + 32] = f2bf(o[qs][2][v] * inv);
      Ob[base + 48] = f2bf(o[qs][3][v] * inv);
    }
}

// ---------------------------------------------------------------------------
// Output projection: out = O @ Wo + bo, fp32 result. 64x64 tile per wave.
// ---------------------------------------------------------------------------
__global__ void __launch_bounds__(256, 1) out_gemm(
    const unsigned short* __restrict__ Ob, const unsigned short* __restrict__ woT,
    const float* __restrict__ bo, float* __restrict__ out)
{
  const unsigned lane = threadIdx.x & 31u, wid = threadIdx.x >> 5;
  const unsigned r = lane & 15u, hi = lane >> 4;
  const int m0 = (blockIdx.y * 8 + wid) * 64;
  const int n0 = blockIdx.x * 64;

  v8f acc[4][4] = {};
  gemm_64x64(Ob, woT, m0, n0, r, hi, acc);

  float bs[4];
#pragma unroll
  for (int j = 0; j < 4; ++j) bs[j] = bo[n0 + j * 16 + r];
#pragma unroll
  for (int i = 0; i < 4; ++i)
    for (int j = 0; j < 4; ++j)
      for (int v = 0; v < 8; ++v)
        out[(size_t)(m0 + i * 16 + v + 8 * hi) * EMBED + n0 + j * 16 + r] =
            acc[i][j][v] + bs[j];
}

// ---------------------------------------------------------------------------
// Host launcher
// ---------------------------------------------------------------------------
extern "C" void kernel_launch(void* const* d_in, const int* in_sizes, int n_in,
                              void* d_out, int out_size, void* d_ws, size_t ws_size,
                              hipStream_t stream) {
  const float* x  = (const float*)d_in[0];
  const float* Wq = (const float*)d_in[1];
  const float* bq = (const float*)d_in[2];
  const float* Wk = (const float*)d_in[3];
  const float* bk = (const float*)d_in[4];
  const float* Wv = (const float*)d_in[5];
  const float* bv = (const float*)d_in[6];
  const float* Wo = (const float*)d_in[7];
  const float* bo = (const float*)d_in[8];
  float* out = (float*)d_out;

  unsigned char* ws = (unsigned char*)d_ws;
  unsigned short* xb  = (unsigned short*)(ws);                    // 16 MiB
  unsigned short* wqT = (unsigned short*)(ws + (16ull << 20));    //  2 MiB
  unsigned short* wkT = (unsigned short*)(ws + (18ull << 20));    //  2 MiB
  unsigned short* wvT = (unsigned short*)(ws + (20ull << 20));    //  2 MiB
  unsigned short* woT = (unsigned short*)(ws + (22ull << 20));    //  2 MiB
  unsigned short* Qb  = (unsigned short*)(ws + (24ull << 20));    // 16 MiB
  unsigned short* Kb  = (unsigned short*)(ws + (40ull << 20));    // 16 MiB
  unsigned short* Vt  = (unsigned short*)(ws + (56ull << 20));    // 16 MiB
  unsigned short* Ob  = (unsigned short*)(ws + (72ull << 20));    // 16 MiB

  const int nX = MROWS * EMBED;  // 8,388,608
  cvt_f32_to_bf16<<<nX / 256, 256, 0, stream>>>(x, xb, nX);
  cvt_transpose_bf16<<<(EMBED * EMBED) / 256, 256, 0, stream>>>(Wq, wqT);
  cvt_transpose_bf16<<<(EMBED * EMBED) / 256, 256, 0, stream>>>(Wk, wkT);
  cvt_transpose_bf16<<<(EMBED * EMBED) / 256, 256, 0, stream>>>(Wv, wvT);
  cvt_transpose_bf16<<<(EMBED * EMBED) / 256, 256, 0, stream>>>(Wo, woT);

  // 128 M-tiles (8 per block) x 16 N-tiles, 3 projections
  qkv_gemm<<<dim3(16, 16, 3), 256, 0, stream>>>(xb, wqT, wkT, wvT,
                                                bq, bk, bv, Qb, Kb, Vt);
  // 4 * 16 * 64 = 4096 waves, 8 per block
  flash_attn<<<512, 256, 0, stream>>>(Qb, Kb, Vt, Ob);
  out_gemm<<<dim3(16, 16), 256, 0, stream>>>(Ob, woT, bo, out);
}
